// QCRModule_4260607558266
// MI455X (gfx1250) — matrix operand order repaired and checked by reading, not compiled, not run on hardware
//
#include <hip/hip_runtime.h>
#include <hip/hip_bf16.h>

// ---------------------------------------------------------------------------
// Problem constants (from the reference)
// ---------------------------------------------------------------------------
constexpr int Bq  = 64;     // batch
constexpr int Rr  = 32;     // image regions
constexpr int Dd  = 512;    // model dim
constexpr int Tt  = 10;     // caption turns
constexpr int Hh  = 8;      // heads
constexpr int DHd = 64;     // head dim
constexpr int FFd = 2048;   // ffn dim
constexpr int Ll  = 6;      // layers
constexpr float NEGV = -1.0e10f;

typedef __attribute__((ext_vector_type(16))) __bf16 v16bf;
typedef __attribute__((ext_vector_type(8)))  __bf16 v8bf;
typedef __attribute__((ext_vector_type(8)))  float  v8f;

union FragU { v16bf v; v8bf h[2]; };

static __device__ __forceinline__ v8f vzero8() {
    v8f z;
#pragma unroll
    for (int i = 0; i < 8; ++i) z[i] = 0.0f;
    return z;
}

// ---------------------------------------------------------------------------
// WMMA bf16 GEMM:  Y[M,N] = act( A[M,K] * W[N,K]^T + bias (+ res) )
// A, W are bf16; res/bias f32; output f32 or bf16 (OBF).
// EPI: 0 = bias, 1 = bias+relu, 2 = bias+residual, 3 = no bias, leaky+clamp.
// Block = 128 threads (4 waves). Wave tile 32(M) x 64(N); block tile 64 x 128.
// Requires M%64==0, N%128==0, K%32==0 (holds for every GEMM here).
// ---------------------------------------------------------------------------
template <int EPI, bool OBF>
__global__ __launch_bounds__(128)
void gemm_wmma(const __bf16* __restrict__ A, const __bf16* __restrict__ W,
               const float* __restrict__ bias, const float* __restrict__ res,
               float* __restrict__ Yf, __bf16* __restrict__ Yb,
               int M, int N, int Kd) {
    const int lane    = threadIdx.x & 31;
    const int wv      = threadIdx.x >> 5;
    const int wm      = wv & 1;
    const int wn      = wv >> 1;
    const int rowBase = blockIdx.y * 64 + wm * 32;
    const int colBase = blockIdx.x * 128 + wn * 64;
    const int halfSel = lane >> 4;      // 0: lanes 0-15, 1: lanes 16-31
    const int l16     = lane & 15;

    v8f acc[2][4];
#pragma unroll
    for (int mt = 0; mt < 2; ++mt)
#pragma unroll
        for (int nt = 0; nt < 4; ++nt) acc[mt][nt] = vzero8();

    // A fragment: per lane 16 contiguous K values starting at halfSel*16 (two 8-elem chunks).
    const __bf16* __restrict__ A0 = A + (size_t)(rowBase + l16) * Kd + halfSel * 16;
    const __bf16* __restrict__ A1 = A0 + (size_t)16 * Kd;
    // B fragment: per lane 16 contiguous K values starting at kb2 = halfSel*16.
    const __bf16* __restrict__ Wb0 = W + (size_t)(colBase + l16) * Kd + halfSel * 16;

    for (int k0 = 0; k0 < Kd; k0 += 32) {
        FragU a0, a1;
        a0.h[0] = *(const v8bf*)(A0 + k0);
        a0.h[1] = *(const v8bf*)(A0 + k0 + 8);
        a1.h[0] = *(const v8bf*)(A1 + k0);
        a1.h[1] = *(const v8bf*)(A1 + k0 + 8);
#pragma unroll
        for (int nt = 0; nt < 4; ++nt) {
            const v16bf bb = *(const v16bf*)(Wb0 + (size_t)nt * 16 * Kd + k0);
            acc[0][nt] = __builtin_amdgcn_wmma_f32_16x16x32_bf16(
                false, a0.v, false, bb, (short)0, acc[0][nt], false, false);
            acc[1][nt] = __builtin_amdgcn_wmma_f32_16x16x32_bf16(
                false, a1.v, false, bb, (short)0, acc[1][nt], false, false);
        }
    }

    // Epilogue. C/D layout: lane holds col l16; VGPR v holds row v + halfSel*8.
#pragma unroll
    for (int mt = 0; mt < 2; ++mt) {
#pragma unroll
        for (int nt = 0; nt < 4; ++nt) {
            const int col = colBase + nt * 16 + l16;
            const float bval = (EPI == 3) ? 0.0f : bias[col];
#pragma unroll
            for (int v = 0; v < 8; ++v) {
                const int row = rowBase + mt * 16 + v + halfSel * 8;
                float y = acc[mt][nt][v] + bval;
                if (EPI == 2) y += res[(size_t)row * N + col];
                if (EPI == 1) y = fmaxf(y, 0.0f);
                if (EPI == 3) { y = (y > 0.0f) ? y : 0.1f * y; y = fmaxf(y, NEGV); }
                if (OBF) Yb[(size_t)row * N + col] = (__bf16)y;
                else     Yf[(size_t)row * N + col] = y;
            }
        }
    }
}

// ---------------------------------------------------------------------------
// f32 -> bf16 conversion (weights, one pass per call)
// ---------------------------------------------------------------------------
__global__ void cvt_bf16(const float* __restrict__ src, __bf16* __restrict__ dst, int n) {
    const int i = blockIdx.x * blockDim.x + threadIdx.x;
    if (i < n) dst[i] = (__bf16)src[i];
}

// ---------------------------------------------------------------------------
// Fused attention: one wave per (b, h, q). S <= 42, DH = 64 (2 floats/lane).
// Output written as bf16 (consumed only by the Wo GEMM).
// ---------------------------------------------------------------------------
__global__ __launch_bounds__(128)
void attn_kernel(const float* __restrict__ Q, const float* __restrict__ K,
                 const float* __restrict__ V, const unsigned char* __restrict__ tmask,
                 __bf16* __restrict__ O, int S) {
    const int wid  = (blockIdx.x * blockDim.x + threadIdx.x) >> 5;
    const int lane = threadIdx.x & 31;
    const int total = Bq * Hh * S;
    if (wid >= total) return;
    const int q = wid % S;
    const int h = (wid / S) % Hh;
    const int b = wid / (S * Hh);

    const float* qp = Q + (size_t)(b * S + q) * Dd + h * DHd;
    const float q0 = qp[2 * lane], q1 = qp[2 * lane + 1];

    float sc0 = -3.0e38f, sc1 = -3.0e38f;   // scores for keys lane, lane+32
    for (int s = 0; s < S; ++s) {
        const float* kp = K + (size_t)(b * S + s) * Dd + h * DHd;
        float part = q0 * kp[2 * lane] + q1 * kp[2 * lane + 1];
#pragma unroll
        for (int off = 16; off; off >>= 1) part += __shfl_xor(part, off, 32);
        float sco = part * 0.125f;          // 1/sqrt(64)
        if (tmask != nullptr && s >= Rr && tmask[b * Tt + (s - Rr)]) sco = NEGV;
        if (s < 32) { if (lane == s)      sc0 = sco; }
        else        { if (lane == s - 32) sc1 = sco; }
    }

    float m = fmaxf(sc0, sc1);
#pragma unroll
    for (int off = 16; off; off >>= 1) m = fmaxf(m, __shfl_xor(m, off, 32));
    const float e0 = __expf(sc0 - m), e1 = __expf(sc1 - m);
    float sum = e0 + e1;
#pragma unroll
    for (int off = 16; off; off >>= 1) sum += __shfl_xor(sum, off, 32);
    const float inv = 1.0f / sum;
    const float a0 = e0 * inv, a1 = e1 * inv;

    float acc0 = 0.0f, acc1 = 0.0f;
    for (int s = 0; s < S; ++s) {
        const float asel = (s < 32) ? a0 : a1;
        const float a = __shfl(asel, s & 31, 32);
        const float* vp = V + (size_t)(b * S + s) * Dd + h * DHd;
        acc0 += a * vp[2 * lane];
        acc1 += a * vp[2 * lane + 1];
    }
    __bf16* op = O + (size_t)(b * S + q) * Dd + h * DHd;
    op[2 * lane]     = (__bf16)acc0;
    op[2 * lane + 1] = (__bf16)acc1;
}

// ---------------------------------------------------------------------------
// LayerNorm over last dim (512). One wave per row.
// Writes f32 (residual path) and bf16 (next GEMM's A operand).
// ---------------------------------------------------------------------------
__global__ __launch_bounds__(128)
void ln_kernel(const float* __restrict__ X, const float* __restrict__ w,
               const float* __restrict__ bb, float* __restrict__ Y,
               __bf16* __restrict__ Ybf, int M) {
    const int row  = (blockIdx.x * blockDim.x + threadIdx.x) >> 5;
    const int lane = threadIdx.x & 31;
    if (row >= M) return;
    const float* xp = X + (size_t)row * Dd;
    float xv[16];
    float s = 0.0f;
#pragma unroll
    for (int i = 0; i < 16; ++i) { xv[i] = xp[lane + 32 * i]; s += xv[i]; }
#pragma unroll
    for (int off = 16; off; off >>= 1) s += __shfl_xor(s, off, 32);
    const float mean = s * (1.0f / 512.0f);
    float v = 0.0f;
#pragma unroll
    for (int i = 0; i < 16; ++i) { const float d = xv[i] - mean; v += d * d; }
#pragma unroll
    for (int off = 16; off; off >>= 1) v += __shfl_xor(v, off, 32);
    const float inv = rsqrtf(v * (1.0f / 512.0f) + 1e-5f);
    float* yp = Y + (size_t)row * Dd;
    __bf16* ybp = Ybf + (size_t)row * Dd;
#pragma unroll
    for (int i = 0; i < 16; ++i) {
        const int c = lane + 32 * i;
        const float yv = (xv[i] - mean) * inv * w[c] + bb[c];
        yp[c]  = yv;
        ybp[c] = (__bf16)yv;
    }
}

// ---------------------------------------------------------------------------
// _turn_sim tail: causal softmax-weighted sums + mean over R.
// 1 block of 32 threads per (b_cap, b_img); lane = region r.
// sim layout: sim[n, bcap*T + t]  with n = b_img*R + r.
// ---------------------------------------------------------------------------
__global__ void turnsim_kernel(const float* __restrict__ sim, float* __restrict__ out) {
    const int bcap = blockIdx.x / Bq;
    const int bimg = blockIdx.x % Bq;
    const int r    = threadIdx.x;           // 0..31
    const int n    = bimg * Rr + r;
    float s[Tt];
#pragma unroll
    for (int t = 0; t < Tt; ++t)
        s[t] = sim[(size_t)n * (Bq * Tt) + bcap * Tt + t];
#pragma unroll
    for (int t = 0; t < Tt; ++t) {
        float m = -3.0e38f;
        for (int tp = 0; tp <= t; ++tp) m = fmaxf(m, s[tp]);
        float den = 0.0f, num = 0.0f;
        for (int tp = 0; tp <= t; ++tp) {
            const float e = __expf(s[tp] - m);
            den += e; num += e * s[tp];
        }
        float wv = num / den;
#pragma unroll
        for (int off = 16; off; off >>= 1) wv += __shfl_xor(wv, off, 32);
        if (r == 0)
            out[(size_t)bcap * (Bq * Tt) + bimg * Tt + t] = wv * (1.0f / 32.0f);
    }
}

// ---------------------------------------------------------------------------
// Data movement / assembly (each writes f32 + bf16 copies where needed)
// ---------------------------------------------------------------------------
__global__ void gather_capsel(const float* __restrict__ cap, const int* __restrict__ idx,
                              __bf16* __restrict__ dst) {
    const int i = blockIdx.x * blockDim.x + threadIdx.x;
    if (i >= Bq * Tt * Dd) return;
    const int d = i % Dd, row = i / Dd;
    const int b = row / Tt, t = row % Tt;
    dst[i] = (__bf16)cap[((size_t)b * Tt + idx[t]) * Dd + d];
}

__global__ void build_src1(const float* __restrict__ gf, const float* __restrict__ img,
                           float* __restrict__ X, __bf16* __restrict__ Xbf) {
    const int i = blockIdx.x * blockDim.x + threadIdx.x;
    if (i >= Bq * 33 * Dd) return;
    const int d = i % Dd, row = i / Dd;
    const int b = row / 33, p = row % 33;
    const float v = (p == 0) ? gf[(size_t)b * Dd + d]
                             : img[((size_t)b * Rr + (p - 1)) * Dd + d];
    X[i] = v; Xbf[i] = (__bf16)v;
}

__global__ void extract_img(const float* __restrict__ X, float* __restrict__ dstF,
                            __bf16* __restrict__ dstB, int S, int start) {
    const int i = blockIdx.x * blockDim.x + threadIdx.x;
    if (i >= Bq * Rr * Dd) return;
    const int d = i % Dd, nrow = i / Dd;
    const int b = nrow / Rr, r = nrow % Rr;
    const float v = X[((size_t)b * S + start + r) * Dd + d];
    dstF[i] = v; dstB[i] = (__bf16)v;
}

__global__ void build_src2(const float* __restrict__ img1, const float* __restrict__ cap,
                           float* __restrict__ X, __bf16* __restrict__ Xbf) {
    const int i = blockIdx.x * blockDim.x + threadIdx.x;
    if (i >= Bq * 42 * Dd) return;
    const int d = i % Dd, row = i / Dd;
    const int b = row / 42, p = row % 42;
    const float v = (p < Rr) ? img1[((size_t)b * Rr + p) * Dd + d]
                             : cap[((size_t)b * Tt + (p - Rr)) * Dd + d];
    X[i] = v; Xbf[i] = (__bf16)v;
}

// ---------------------------------------------------------------------------
// Host orchestration
// ---------------------------------------------------------------------------
extern "C" void kernel_launch(void* const* d_in, const int* in_sizes, int n_in,
                              void* d_out, int out_size, void* d_ws, size_t ws_size,
                              hipStream_t stream) {
    const float* img_emb = (const float*)d_in[0];
    const float* cap_emb = (const float*)d_in[1];
    const float* gfeat   = (const float*)d_in[2];
    const float* Wq = (const float*)d_in[3];  const float* bq = (const float*)d_in[4];
    const float* Wk = (const float*)d_in[5];  const float* bk = (const float*)d_in[6];
    const float* Wv = (const float*)d_in[7];  const float* bv = (const float*)d_in[8];
    const float* Wo = (const float*)d_in[9];  const float* bo = (const float*)d_in[10];
    const float* ln1w = (const float*)d_in[11]; const float* ln1b = (const float*)d_in[12];
    const float* W1 = (const float*)d_in[13]; const float* b1 = (const float*)d_in[14];
    const float* W2 = (const float*)d_in[15]; const float* b2 = (const float*)d_in[16];
    const float* ln2w = (const float*)d_in[17]; const float* ln2b = (const float*)d_in[18];
    const unsigned char* t_mask = (const unsigned char*)d_in[19];
    const int* indices = (const int*)d_in[20];
    float* out = (float*)d_out;

    // ---- workspace carve-out (256B aligned blocks) ----
    char* wp = (char*)d_ws;
    auto carve = [&](size_t bytes) -> char* {
        char* p = wp; wp += (bytes + 255) & ~(size_t)255; return p;
    };
    const size_t MMAX = (size_t)Bq * 42;                    // 2688 rows max
    float*  Xf    = (float*)carve(MMAX * Dd * 4);
    __bf16* Xbf   = (__bf16*)carve(MMAX * Dd * 2);
    float*  Qf    = (float*)carve(MMAX * Dd * 4);
    float*  Kf    = (float*)carve(MMAX * Dd * 4);
    float*  Vf    = (float*)carve(MMAX * Dd * 4);
    __bf16* Obf   = (__bf16*)carve(MMAX * Dd * 2);
    float*  Pf    = (float*)carve(MMAX * Dd * 4);
    __bf16* FFHb  = (__bf16*)carve(MMAX * FFd * 2);
    __bf16* WqB   = (__bf16*)carve((size_t)Ll * Dd * Dd * 2);
    __bf16* WkB   = (__bf16*)carve((size_t)Ll * Dd * Dd * 2);
    __bf16* WvB   = (__bf16*)carve((size_t)Ll * Dd * Dd * 2);
    __bf16* WoB   = (__bf16*)carve((size_t)Ll * Dd * Dd * 2);
    __bf16* W1B   = (__bf16*)carve((size_t)Ll * FFd * Dd * 2);
    __bf16* W2B   = (__bf16*)carve((size_t)Ll * Dd * FFd * 2);
    __bf16* capB  = (__bf16*)carve((size_t)Bq * Tt * Dd * 2);
    float*  imgF  = (float*)carve((size_t)Bq * Rr * Dd * 4);
    __bf16* imgB  = (__bf16*)carve((size_t)Bq * Rr * Dd * 2);
    float*  simF  = (float*)carve((size_t)(Bq * Rr) * (Bq * Tt) * 4);

    const dim3 blk(128);

    // ---- one-shot weight conversion to bf16 ----
    {
        const int nDD = Ll * Dd * Dd, nFD = Ll * FFd * Dd;
        cvt_bf16<<<(nDD + 255) / 256, 256, 0, stream>>>(Wq, WqB, nDD);
        cvt_bf16<<<(nDD + 255) / 256, 256, 0, stream>>>(Wk, WkB, nDD);
        cvt_bf16<<<(nDD + 255) / 256, 256, 0, stream>>>(Wv, WvB, nDD);
        cvt_bf16<<<(nDD + 255) / 256, 256, 0, stream>>>(Wo, WoB, nDD);
        cvt_bf16<<<(nFD + 255) / 256, 256, 0, stream>>>(W1, W1B, nFD);
        cvt_bf16<<<(nFD + 255) / 256, 256, 0, stream>>>(W2, W2B, nFD);
    }

    auto run_stack = [&](int S, const unsigned char* mask) {
        const int M = Bq * S;
        const dim3 gD(Dd / 128, M / 64);
        const dim3 gF(FFd / 128, M / 64);
        const int attnBlocks = (Bq * Hh * S + 3) / 4;
        const int lnBlocks   = (M + 3) / 4;
        for (int l = 0; l < Ll; ++l) {
            const __bf16* wq = WqB + (size_t)l * Dd * Dd;
            const __bf16* wk = WkB + (size_t)l * Dd * Dd;
            const __bf16* wv = WvB + (size_t)l * Dd * Dd;
            const __bf16* wo = WoB + (size_t)l * Dd * Dd;
            const __bf16* w1 = W1B + (size_t)l * FFd * Dd;
            const __bf16* w2 = W2B + (size_t)l * Dd * FFd;

            gemm_wmma<0, false><<<gD, blk, 0, stream>>>(Xbf, wq, bq + (size_t)l * Dd, nullptr, Qf, nullptr, M, Dd, Dd);
            gemm_wmma<0, false><<<gD, blk, 0, stream>>>(Xbf, wk, bk + (size_t)l * Dd, nullptr, Kf, nullptr, M, Dd, Dd);
            gemm_wmma<0, false><<<gD, blk, 0, stream>>>(Xbf, wv, bv + (size_t)l * Dd, nullptr, Vf, nullptr, M, Dd, Dd);
            attn_kernel<<<attnBlocks, blk, 0, stream>>>(Qf, Kf, Vf, mask, Obf, S);
            gemm_wmma<2, false><<<gD, blk, 0, stream>>>(Obf, wo, bo + (size_t)l * Dd, Xf, Pf, nullptr, M, Dd, Dd);
            ln_kernel<<<lnBlocks, blk, 0, stream>>>(Pf, ln1w + (size_t)l * Dd, ln1b + (size_t)l * Dd, Xf, Xbf, M);
            gemm_wmma<1, true ><<<gF, blk, 0, stream>>>(Xbf, w1, b1 + (size_t)l * FFd, nullptr, nullptr, FFHb, M, FFd, Dd);
            gemm_wmma<2, false><<<gD, blk, 0, stream>>>(FFHb, w2, b2 + (size_t)l * Dd, Xf, Pf, nullptr, M, Dd, FFd);
            ln_kernel<<<lnBlocks, blk, 0, stream>>>(Pf, ln2w + (size_t)l * Dd, ln2b + (size_t)l * Dd, Xf, Xbf, M);
        }
    };

    // cap_sel = take(cap_emb, indices, axis=1), bf16
    gather_capsel<<<(Bq * Tt * Dd + 255) / 256, 256, 0, stream>>>(cap_emb, indices, capB);

    // ---- stage 1: [global ; img] encoder, S = 33 ----
    build_src1<<<(Bq * 33 * Dd + 255) / 256, 256, 0, stream>>>(gfeat, img_emb, Xf, Xbf);
    run_stack(33, nullptr);
    extract_img<<<(Bq * Rr * Dd + 255) / 256, 256, 0, stream>>>(Xf, imgF, imgB, 33, 1);

    // sim_local_1: (2048 x 512) x (640 x 512)^T with leaky + clamp epilogue
    gemm_wmma<3, false><<<dim3((Bq * Tt) / 128, (Bq * Rr) / 64), blk, 0, stream>>>(
        imgB, capB, nullptr, nullptr, simF, nullptr, Bq * Rr, Bq * Tt, Dd);
    turnsim_kernel<<<Bq * Bq, 32, 0, stream>>>(simF, out);

    // ---- stage 2: [img1 ; cap] encoder, S = 42, masked ----
    build_src2<<<(Bq * 42 * Dd + 255) / 256, 256, 0, stream>>>(imgF, cap_emb, Xf, Xbf);
    run_stack(42, t_mask);
    extract_img<<<(Bq * Rr * Dd + 255) / 256, 256, 0, stream>>>(Xf, imgF, imgB, 42, 0);

    gemm_wmma<3, false><<<dim3((Bq * Tt) / 128, (Bq * Rr) / 64), blk, 0, stream>>>(
        imgB, capB, nullptr, nullptr, simF, nullptr, Bq * Rr, Bq * Tt, Dd);
    turnsim_kernel<<<Bq * Bq, 32, 0, stream>>>(simF, out + (size_t)Bq * Bq * Tt);
}
